// PointNet2CornerDetection_24919400252187
// MI455X (gfx1250) — compile-verified, matrix operand order checked by reading, not run on hardware
//
#include <hip/hip_runtime.h>
#include <cstdint>
#include <cstddef>

typedef unsigned short u16;
typedef __attribute__((ext_vector_type(16))) __bf16 bf16x16;
typedef __attribute__((ext_vector_type(8)))  float  f32x8;
typedef __attribute__((ext_vector_type(4)))  unsigned int u32x4;
typedef __attribute__((ext_vector_type(4)))  unsigned int v4u;
typedef __attribute__((ext_vector_type(8)))  int v8i;
typedef __attribute__((ext_vector_type(4)))  int v4i;

__device__ __forceinline__ u16 f2bf(float f) {
  unsigned int u = __builtin_bit_cast(unsigned int, f);
  u += 0x7fffu + ((u >> 16) & 1u);   // round-to-nearest-even
  return (u16)(u >> 16);
}

// ---------------------------------------------------------------------------
// Fold BN into weights: WH[co][ciP] = bf16(W*scale), bias = (b-mean)*scale+beta
// ---------------------------------------------------------------------------
__global__ void fold_weights_k(const float* __restrict__ W, const float* __restrict__ b,
                               const float* __restrict__ g, const float* __restrict__ be,
                               const float* __restrict__ mn, const float* __restrict__ vr,
                               u16* __restrict__ WH, float* __restrict__ biasF,
                               int cout, int cin, int cinP) {
  int total = cout * cinP;
  for (int t = blockIdx.x * blockDim.x + threadIdx.x; t < total; t += gridDim.x * blockDim.x) {
    int ci = t % cinP, co = t / cinP;
    float sc = g[co] / sqrtf(vr[co] + 1e-5f);
    float v = (ci < cin) ? W[(size_t)co * cin + ci] * sc : 0.f;
    WH[t] = f2bf(v);
    if (ci == 0) biasF[co] = (b[co] - mn[co]) * sc + be[co];
  }
}

// ---------------------------------------------------------------------------
// WMMA GEMM: D = relu(A*X + bias). A [coutP][cinP] bf16, X [M][cinP] bf16.
// Block = 4 waves; the block's 16-row A tile is DMA'd to LDS by the Tensor
// Data Mover once, then each wave computes a 16x64 strip (4 WMMAs / A-frag).
// outF (optional) f32 [cout][M]; outH (optional) bf16 [M][coutP].
// ---------------------------------------------------------------------------
__global__ __launch_bounds__(128)
void gemm_bn_relu_k(const u16* __restrict__ A, const u16* __restrict__ X,
                    const float* __restrict__ bias,
                    float* __restrict__ outF, u16* __restrict__ outH,
                    int M, int cout, int coutP, int cinP) {
  __shared__ u16 As[16 * 1824];          // max cinP = 1792 (+ pad)
  const int lane = threadIdx.x & 31;
  const int wave = threadIdx.x >> 5;
  const int rowBase = blockIdx.y * 16;
  const int r = lane & 15;
  const int half = lane >> 4;

  // ---- TDM: stage A[rowBase .. rowBase+15][0 .. cinP) into LDS ----
  if (wave == 0) {
    unsigned ldsOff = (unsigned)(size_t)(void*)&As[0];
    unsigned long long ga = (unsigned long long)(size_t)(A + (size_t)rowBase * cinP);
    unsigned d0 = (unsigned)cinP;            // elements per row (data_size units)
    v4u g0;
    g0[0] = 1u;                              // count = 1 (valid descriptor)
    g0[1] = ldsOff;                          // lds_addr
    g0[2] = (unsigned)ga;                    // global_addr[95:64]
    g0[3] = (unsigned)((ga >> 32) & 0x01ffffffull) | 0x80000000u; // addr hi | type=2
    v8i g1;
    g1[0] = (int)(1u << 16);                 // data_size = 1 -> 2 bytes
    g1[1] = (int)((d0 & 0xffffu) << 16);     // tensor_dim0[15:0]
    g1[2] = (int)(((d0 >> 16) & 0xffffu) | (16u << 16)); // dim0 hi | tensor_dim1 lo
    g1[3] = (int)((d0 & 0xffffu) << 16);     // dim1 hi=0 | tile_dim0 = cinP
    g1[4] = 16;                              // tile_dim1 = 16, tile_dim2 = 0
    g1[5] = (int)d0;                         // tensor_dim0_stride[31:0]
    g1[6] = 0;                               // stride0 hi | stride1 lo
    g1[7] = 0;
    v4i gz = (v4i)0;
    v8i gz8 = (v8i)0;
    __builtin_amdgcn_tensor_load_to_lds(g0, g1, gz, gz, gz8, 0);
    __builtin_amdgcn_s_wait_tensorcnt(0);
  }
  __syncthreads();

  const u16* Arow = &As[0] + (size_t)r * cinP + half * 8;
  const int c0 = blockIdx.x * 256 + wave * 64 + r;
  const u16* xp[4];
#pragma unroll
  for (int s = 0; s < 4; ++s) {
    int c = c0 + s * 16;
    if (c >= M) c = M - 1;                   // clamp: keep loads in-bounds, no divergence
    xp[s] = X + (size_t)c * cinP + half * 8;
  }
  f32x8 acc[4];
#pragma unroll
  for (int s = 0; s < 4; ++s) acc[s] = (f32x8)0.0f;

  for (int k0 = 0; k0 < cinP; k0 += 32) {
    union U { bf16x16 v; u32x4 q[2]; } a, bm[4];
    a.q[0] = *(const u32x4*)(Arow + k0);         // K = k0 + half*8 + [0..7]
    a.q[1] = *(const u32x4*)(Arow + k0 + 16);    // K = k0 + 16 + half*8 + [0..7]
#pragma unroll
    for (int s = 0; s < 4; ++s) {
      bm[s].q[0] = *(const u32x4*)(xp[s] + k0);
      bm[s].q[1] = *(const u32x4*)(xp[s] + k0 + 16);
    }
#pragma unroll
    for (int s = 0; s < 4; ++s)
      acc[s] = __builtin_amdgcn_wmma_f32_16x16x32_bf16(false, a.v, false, bm[s].v,
                                                       (short)0, acc[s], false, false);
  }

#pragma unroll
  for (int s = 0; s < 4; ++s) {
    int n = c0 + s * 16;
    if (n < M) {
#pragma unroll
      for (int j = 0; j < 8; ++j) {
        int m = rowBase + j + half * 8;
        float y = acc[s][j] + bias[m];
        y = y > 0.f ? y : 0.f;
        if (outH) outH[(size_t)n * coutP + m] = f2bf(y);
        if (outF && m < cout) outF[(size_t)m * M + n] = y;
      }
    }
  }
}

// ---------------------------------------------------------------------------
// Farthest point sampling: one block per batch; exact argmax tie -> lowest idx
// ---------------------------------------------------------------------------
__global__ void fps_k(const float* __restrict__ xyz, float* __restrict__ distws,
                      int* __restrict__ outIdx, int N, int S) {
  int b = blockIdx.x;
  const float* px = xyz + (size_t)b * N * 3;
  float* dist = distws + (size_t)b * N;
  __shared__ float sv[256];
  __shared__ int   si[256];
  __shared__ int   sfar;
  for (int i = threadIdx.x; i < N; i += 256) dist[i] = 1e10f;
  if (threadIdx.x == 0) sfar = 0;
  __syncthreads();
  for (int it = 0; it < S; ++it) {
    int far = sfar;
    if (threadIdx.x == 0) outIdx[b * S + it] = far;
    float cx = px[far * 3 + 0], cy = px[far * 3 + 1], cz = px[far * 3 + 2];
    float best = -1.f; int bidx = 0x7fffffff;
    for (int i = threadIdx.x; i < N; i += 256) {
      float dx = px[i * 3 + 0] - cx, dy = px[i * 3 + 1] - cy, dz = px[i * 3 + 2] - cz;
      float d = dx * dx + dy * dy + dz * dz;
      float dd = dist[i];
      if (d < dd) { dd = d; dist[i] = dd; }
      if (dd > best) { best = dd; bidx = i; }
    }
    sv[threadIdx.x] = best; si[threadIdx.x] = bidx;
    __syncthreads();
    for (int s2 = 128; s2 > 0; s2 >>= 1) {
      if (threadIdx.x < s2) {
        float ov = sv[threadIdx.x + s2]; int oi = si[threadIdx.x + s2];
        if (ov > sv[threadIdx.x] || (ov == sv[threadIdx.x] && oi < si[threadIdx.x])) {
          sv[threadIdx.x] = ov; si[threadIdx.x] = oi;
        }
      }
      __syncthreads();
    }
    if (threadIdx.x == 0) sfar = si[0];
    __syncthreads();
  }
}

__global__ void gather_xyz_k(const float* __restrict__ xyz, const int* __restrict__ idx,
                             float* __restrict__ nxyz, int N, int S, int B) {
  int t = blockIdx.x * blockDim.x + threadIdx.x;
  if (t >= B * S) return;
  int b = t / S;
  int j = idx[t];
  nxyz[t * 3 + 0] = xyz[((size_t)b * N + j) * 3 + 0];
  nxyz[t * 3 + 1] = xyz[((size_t)b * N + j) * 3 + 1];
  nxyz[t * 3 + 2] = xyz[((size_t)b * N + j) * 3 + 2];
}

// Ball query: first K in-radius indices (ascending), padded with the first hit.
__global__ void ballquery_k(const float* __restrict__ xyz, const float* __restrict__ nxyz,
                            int* __restrict__ gidx, int N, int S, int K, float r2, int B) {
  int t = blockIdx.x * blockDim.x + threadIdx.x;
  if (t >= B * S) return;
  int b = t / S;
  float cx = nxyz[t * 3 + 0], cy = nxyz[t * 3 + 1], cz = nxyz[t * 3 + 2];
  const float* px = xyz + (size_t)b * N * 3;
  int cnt = 0, first = 0;
  for (int j = 0; j < N && cnt < K; ++j) {
    __builtin_prefetch(px + (j + 96) * 3, 0, 1);
    float dx = px[j * 3 + 0] - cx, dy = px[j * 3 + 1] - cy, dz = px[j * 3 + 2] - cz;
    float d = dx * dx + dy * dy + dz * dz;
    if (d <= r2) {
      if (cnt == 0) first = j;
      gidx[(size_t)t * K + cnt] = j;
      ++cnt;
    }
  }
  for (; cnt < K; ++cnt) gidx[(size_t)t * K + cnt] = first;
}

// Build grouped GEMM input: X [M=B*S*K][cinP] bf16, channels per reference order.
__global__ void group_k(const float* __restrict__ xyz, const float* __restrict__ nxyz,
                        const float* __restrict__ pts, const int* __restrict__ gidx,
                        u16* __restrict__ X, int N, int S, int K, int Cprev, int cinP,
                        int xyzFirst, int B) {
  size_t total = (size_t)B * S * K * cinP;
  for (size_t t = (size_t)blockIdx.x * blockDim.x + threadIdx.x; t < total;
       t += (size_t)gridDim.x * blockDim.x) {
    int c = (int)(t % cinP);
    size_t m = t / cinP;
    size_t bs = m / K;
    int b = (int)(bs / S);
    int j = gidx[m];
    float v = 0.f;
    if (c < Cprev + 3) {
      bool isX; int d;
      if (xyzFirst) { isX = (c < 3); d = c; }
      else          { isX = (c >= Cprev); d = c - Cprev; }
      if (isX) {
        v = xyz[((size_t)b * N + j) * 3 + d] - nxyz[bs * 3 + d];
      } else {
        int pc = xyzFirst ? (c - 3) : c;
        v = pts[(size_t)pc * ((size_t)B * N) + (size_t)b * N + j];
      }
    }
    X[t] = f2bf(v);
  }
}

// Max over K samples: src f32 [cout][B*S*K] -> dst[chOff+co][B*S]
__global__ void maxpool_k(const float* __restrict__ src, float* __restrict__ dst,
                          int S, int K, int cout, int chOff, int B) {
  int total = cout * B * S;
  int t = blockIdx.x * blockDim.x + threadIdx.x;
  if (t >= total) return;
  int m = t % (B * S);
  int co = t / (B * S);
  const float* p = src + (size_t)co * ((size_t)B * S * K) + (size_t)m * K;
  float mx = p[0];
  for (int k = 1; k < K; ++k) mx = fmaxf(mx, p[k]);
  dst[(size_t)(chOff + co) * ((size_t)B * S) + m] = mx;
}

// 3-NN: exact reference semantics (smallest d, ties -> lower index), w=1/(d+1e-8)
__global__ void knn3_k(const float* __restrict__ x1, const float* __restrict__ x2,
                       int* __restrict__ idx3, float* __restrict__ w3,
                       int N1, int N2, int B) {
  int t = blockIdx.x * blockDim.x + threadIdx.x;
  if (t >= B * N1) return;
  int b = t / N1;
  float cx = x1[t * 3 + 0], cy = x1[t * 3 + 1], cz = x1[t * 3 + 2];
  const float* p2 = x2 + (size_t)b * N2 * 3;
  float d0 = 1e30f, d1 = 1e30f, d2 = 1e30f;
  int i0 = 0, i1 = 0, i2 = 0;
  for (int j = 0; j < N2; ++j) {
    __builtin_prefetch(p2 + (j + 64) * 3, 0, 1);
    float dx = p2[j * 3 + 0] - cx, dy = p2[j * 3 + 1] - cy, dz = p2[j * 3 + 2] - cz;
    float d = dx * dx + dy * dy + dz * dz;
    if (d < d0)      { d2 = d1; i2 = i1; d1 = d0; i1 = i0; d0 = d; i0 = j; }
    else if (d < d1) { d2 = d1; i2 = i1; d1 = d;  i1 = j; }
    else if (d < d2) { d2 = d;  i2 = j; }
  }
  float w0 = 1.f / (d0 + 1e-8f), w1 = 1.f / (d1 + 1e-8f), w2 = 1.f / (d2 + 1e-8f);
  float s = w0 + w1 + w2;
  w3[t * 3 + 0] = w0 / s; w3[t * 3 + 1] = w1 / s; w3[t * 3 + 2] = w2 / s;
  idx3[t * 3 + 0] = i0;   idx3[t * 3 + 1] = i1;   idx3[t * 3 + 2] = i2;
}

// Build FP GEMM input: [pts1 | interp(pts2)] -> X [M=B*N1][cinP] bf16
__global__ void fp_input_k(const float* __restrict__ pts1, const float* __restrict__ pts2,
                           const int* __restrict__ idx3, const float* __restrict__ w3,
                           u16* __restrict__ X, int N1, int N2, int C1, int C2,
                           int cinP, int B) {
  size_t total = (size_t)B * N1 * cinP;
  for (size_t t = (size_t)blockIdx.x * blockDim.x + threadIdx.x; t < total;
       t += (size_t)gridDim.x * blockDim.x) {
    int c = (int)(t % cinP);
    size_t m = t / cinP;
    int b = (int)(m / N1);
    float v = 0.f;
    if (c < C1) {
      v = pts1[(size_t)c * ((size_t)B * N1) + m];
    } else if (c < C1 + C2) {
      int cc = c - C1;
      const float* row = pts2 + (size_t)cc * ((size_t)B * N2);
      size_t base = (size_t)b * N2;
      v = w3[m * 3 + 0] * row[base + idx3[m * 3 + 0]]
        + w3[m * 3 + 1] * row[base + idx3[m * 3 + 1]]
        + w3[m * 3 + 2] * row[base + idx3[m * 3 + 2]];
    }
    X[t] = f2bf(v);
  }
}

// Final 1x64 head: logits[m] = W2 . feat[:,m] + b2
__global__ void head2_k(const float* __restrict__ feat, const float* __restrict__ W2,
                        const float* __restrict__ b2, float* __restrict__ out, int M) {
  int m = blockIdx.x * blockDim.x + threadIdx.x;
  if (m >= M) return;
  float s = b2[0];
  for (int c = 0; c < 64; ++c) s += W2[c] * feat[(size_t)c * M + m];
  out[m] = s;
}

// l0_points: src f32 [128][B*N] -> dst [B][128][N]
__global__ void l0copy_k(const float* __restrict__ src, float* __restrict__ dst, int N, int B) {
  size_t total = (size_t)B * 128 * N;
  for (size_t t = (size_t)blockIdx.x * blockDim.x + threadIdx.x; t < total;
       t += (size_t)gridDim.x * blockDim.x) {
    int n = (int)(t % N);
    size_t bc = t / N;
    int c = (int)(bc % 128);
    int b = (int)(bc / 128);
    dst[t] = src[(size_t)c * ((size_t)B * N) + (size_t)b * N + n];
  }
}

// ---------------------------------------------------------------------------
static inline int cdiv(int a, int b) { return (a + b - 1) / b; }
static inline int rnd32(int x) { return (x + 31) & ~31; }

extern "C" void kernel_launch(void* const* d_in, const int* in_sizes, int n_in,
                              void* d_out, int out_size, void* d_ws, size_t ws_size,
                              hipStream_t stream) {
  (void)in_sizes; (void)n_in; (void)out_size; (void)ws_size;
  const int B = 4, N0 = 16384;
  const float* xyz0 = (const float*)d_in[0];

  char* ws = (char*)d_ws;
  size_t off = 0;
  auto alloc = [&](size_t bytes) -> void* {
    void* p = ws + off;
    off += (bytes + 255) & ~(size_t)255;
    return p;
  };

  // Conv layers in d_in order (each conv: W,b,gamma,beta,mean,var)
  struct Spec { int cin, cout; };
  static const Spec specs[31] = {
    {3,32},{32,32},{32,64},        // sa1 branch1
    {3,64},{64,64},{64,128},       // sa1 branch2
    {195,64},{64,64},{64,128},     // sa2 branch1
    {195,128},{128,128},{128,256}, // sa2 branch2
    {387,128},{128,128},{128,256}, // sa3 branch1
    {387,256},{256,256},{256,512}, // sa3 branch2
    {771,256},{256,512},{512,1024},// sa4
    {1792,512},{512,512},          // fp4
    {896,512},{512,256},           // fp3
    {448,256},{256,128},           // fp2
    {128,128},{128,128},{128,128}, // fp1
    {128,64}                       // head1
  };

  struct Lay { const u16* W; const float* b; int cin, cout, cinP; };
  Lay L[31];
  for (int i = 0; i < 31; ++i) {
    int cin = specs[i].cin, cout = specs[i].cout, cinP = rnd32(cin);
    u16*  WH = (u16*)alloc((size_t)cout * cinP * sizeof(u16));
    float* bf = (float*)alloc((size_t)cout * sizeof(float));
    const float* W  = (const float*)d_in[1 + 6 * i + 0];
    const float* bb = (const float*)d_in[1 + 6 * i + 1];
    const float* gm = (const float*)d_in[1 + 6 * i + 2];
    const float* bt = (const float*)d_in[1 + 6 * i + 3];
    const float* mn = (const float*)d_in[1 + 6 * i + 4];
    const float* vr = (const float*)d_in[1 + 6 * i + 5];
    int total = cout * cinP;
    fold_weights_k<<<cdiv(total, 256), 256, 0, stream>>>(W, bb, gm, bt, mn, vr, WH, bf, cout, cin, cinP);
    L[i] = {WH, bf, cin, cout, cinP};
  }
  const float* w2W = (const float*)d_in[1 + 6 * 31 + 0];
  const float* w2b = (const float*)d_in[1 + 6 * 31 + 1];

  // Arenas
  u16*  XA = (u16*)alloc((size_t)9600000 * sizeof(u16));   // grouped/concat GEMM input
  u16*  PA = (u16*)alloc((size_t)8388608 * sizeof(u16));   // ping
  u16*  PB = (u16*)alloc((size_t)8388608 * sizeof(u16));   // pong
  float* FA = (float*)alloc((size_t)10485760 * sizeof(float)); // SA final f32 (pre-pool)

  float* dist   = (float*)alloc((size_t)B * N0 * sizeof(float));
  int*   fpsIdx = (int*)alloc((size_t)B * 640 * sizeof(int));
  int*   gidx   = (int*)alloc((size_t)B * 640 * 32 * sizeof(int));
  float* l1x = (float*)alloc((size_t)B * 640 * 3 * sizeof(float));
  float* l2x = (float*)alloc((size_t)B * 320 * 3 * sizeof(float));
  float* l3x = (float*)alloc((size_t)B * 160 * 3 * sizeof(float));
  float* l4x = (float*)alloc((size_t)B * 40  * 3 * sizeof(float));
  float* l1p = (float*)alloc((size_t)192  * B * 640 * sizeof(float));
  float* l2p = (float*)alloc((size_t)384  * B * 320 * sizeof(float));
  float* l3p = (float*)alloc((size_t)768  * B * 160 * sizeof(float));
  float* l4p = (float*)alloc((size_t)1024 * B * 40  * sizeof(float));
  float* f4o = (float*)alloc((size_t)512 * B * 160 * sizeof(float));
  float* f3o = (float*)alloc((size_t)256 * B * 320 * sizeof(float));
  float* f2o = (float*)alloc((size_t)128 * B * 640 * sizeof(float));
  float* f1o = (float*)alloc((size_t)128 * B * N0  * sizeof(float));
  u16*  FH   = (u16*)alloc((size_t)B * N0 * 128 * sizeof(u16));
  float* featF = (float*)alloc((size_t)64 * B * N0 * sizeof(float));
  int*   knnI = (int*)alloc((size_t)B * N0 * 3 * sizeof(int));
  float* knnW = (float*)alloc((size_t)B * N0 * 3 * sizeof(float));

  auto runMLP = [&](const u16* X, int M, const Lay* lay, int nl, float* outF, u16* outH) {
    const u16* cur = X;
    for (int i = 0; i < nl; ++i) {
      bool last = (i == nl - 1);
      u16*  oh = last ? outH : ((i & 1) ? PB : PA);
      float* of = last ? outF : nullptr;
      dim3 g(cdiv(M, 256), lay[i].cout / 16);
      gemm_bn_relu_k<<<g, 128, 0, stream>>>(lay[i].W, cur, lay[i].b, of, oh,
                                            M, lay[i].cout, lay[i].cout, lay[i].cinP);
      cur = oh;
    }
  };

  auto saStage = [&](const float* xyzPrev, int Nprev, const float* ptsPrev, int Cprev,
                     int S, float* newXyz, float* dstPts,
                     const float* radii, const int* Ks,
                     const Lay* const* branches, const int* nls, int nbranch, int xyzFirst) {
    fps_k<<<B, 256, 0, stream>>>(xyzPrev, dist, fpsIdx, Nprev, S);
    gather_xyz_k<<<cdiv(B * S, 128), 128, 0, stream>>>(xyzPrev, fpsIdx, newXyz, Nprev, S, B);
    int chOff = 0;
    for (int br = 0; br < nbranch; ++br) {
      int K = Ks[br];
      float r2 = radii[br] * radii[br];
      ballquery_k<<<cdiv(B * S, 128), 128, 0, stream>>>(xyzPrev, newXyz, gidx, Nprev, S, K, r2, B);
      const Lay* lay = branches[br]; int nl = nls[br];
      int cinP = lay[0].cinP;
      long long total = (long long)B * S * K * cinP;
      int blocks = (int)((total + 255) / 256);
      group_k<<<blocks, 256, 0, stream>>>(xyzPrev, newXyz, ptsPrev, gidx, XA,
                                          Nprev, S, K, Cprev, cinP, xyzFirst, B);
      int M = B * S * K;
      runMLP(XA, M, lay, nl, FA, nullptr);
      int coutL = lay[nl - 1].cout;
      maxpool_k<<<cdiv(coutL * B * S, 256), 256, 0, stream>>>(FA, dstPts, S, K, coutL, chOff, B);
      chOff += coutL;
    }
  };

  auto fpStage = [&](const float* x1, int N1, const float* x2, int N2,
                     const float* pts1, int C1, const float* pts2, int C2,
                     const Lay* lay, int nl, float* outF, u16* outH) {
    knn3_k<<<cdiv(B * N1, 128), 128, 0, stream>>>(x1, x2, knnI, knnW, N1, N2, B);
    int cinP = lay[0].cinP;
    long long total = (long long)B * N1 * cinP;
    int blocks = (int)((total + 255) / 256);
    fp_input_k<<<blocks, 256, 0, stream>>>(pts1, pts2, knnI, knnW, XA, N1, N2, C1, C2, cinP, B);
    runMLP(XA, B * N1, lay, nl, outF, outH);
  };

  // ---- SA stages ----
  {
    const float radii[2] = {0.05f, 0.1f}; const int Ks[2] = {16, 32};
    const Lay* br[2] = {&L[0], &L[3]}; const int nls[2] = {3, 3};
    saStage(xyz0, N0, nullptr, 0, 640, l1x, l1p, radii, Ks, br, nls, 2, 0);
  }
  {
    const float radii[2] = {0.1f, 0.2f}; const int Ks[2] = {16, 32};
    const Lay* br[2] = {&L[6], &L[9]}; const int nls[2] = {3, 3};
    saStage(l1x, 640, l1p, 192, 320, l2x, l2p, radii, Ks, br, nls, 2, 0);
  }
  {
    const float radii[2] = {0.2f, 0.4f}; const int Ks[2] = {16, 32};
    const Lay* br[2] = {&L[12], &L[15]}; const int nls[2] = {3, 3};
    saStage(l2x, 320, l2p, 384, 160, l3x, l3p, radii, Ks, br, nls, 2, 0);
  }
  {
    const float radii[1] = {0.4f}; const int Ks[1] = {32};
    const Lay* br[1] = {&L[18]}; const int nls[1] = {3};
    saStage(l3x, 160, l3p, 768, 40, l4x, l4p, radii, Ks, br, nls, 1, /*xyzFirst=*/1);
  }

  // ---- FP stages ----
  fpStage(l3x, 160, l4x, 40,  l3p, 768, l4p, 1024, &L[21], 2, f4o, nullptr);
  fpStage(l2x, 320, l3x, 160, l2p, 384, f4o, 512,  &L[23], 2, f3o, nullptr);
  fpStage(l1x, 640, l2x, 320, l1p, 192, f3o, 256,  &L[25], 2, f2o, nullptr);
  fpStage(xyz0, N0, l1x, 640, nullptr, 0, f2o, 128, &L[27], 3, f1o, FH);

  // ---- Head ----
  runMLP(FH, B * N0, &L[30], 1, featF, nullptr);
  head2_k<<<cdiv(B * N0, 256), 256, 0, stream>>>(featF, w2W, w2b, (float*)d_out, B * N0);
  l0copy_k<<<cdiv(B * 128 * N0, 256), 256, 0, stream>>>(f1o, (float*)d_out + (size_t)B * N0, N0, B);
}
